// VQVAE_15341623181537
// MI455X (gfx1250) — matrix operand order repaired and checked by reading, not compiled, not run on hardware
//
#include <hip/hip_runtime.h>
#include <hip/hip_bf16.h>
#include <math.h>

typedef __attribute__((ext_vector_type(16))) __bf16 v16bf;
typedef __attribute__((ext_vector_type(8)))  float  v8f;

#define ACT_NONE 0
#define ACT_RELU 1
#define ACT_SIG  2

// ---------------------------------------------------------------------------
// WMMA bf16 GEMM:  C[M,N] = act(A[M,K] @ W[K,N] + bias)
// Block = 128 threads (4 waves) -> 64x64 macro-tile.
// REQUIRES: M % 64 == 0, N % 4 == 0 (true at every call site).
// Branch-free: OOB accesses are address-clamped and value-masked (cndmask),
// never EXEC-predicated, so the compiler cannot version/scalarize the loop.
// Per K-step (32):
//   * W slab 32x64: each thread loads a 4(k) x 4(n) block via 4x b128,
//     packs to bf16, stores 4x ds_store_b64 in B-fragment order.
//   * A fragment: two contiguous 8-float runs per lane -> 4x global b128.
//   * 2x ds_load_b128 per B fragment, 4x v_wmma per wave per step.
// ---------------------------------------------------------------------------
__global__ void __launch_bounds__(128)
gemm_wmma_bf16(const float* __restrict__ A, const float* __restrict__ W,
               const float* __restrict__ bias, float* __restrict__ C,
               int M, int N, int K, int ldc, int act, int tilesN)
{
    __shared__ __bf16 sW[32 * 64];   // 4 KB, fragment-order layout

    int tm = blockIdx.x / tilesN;
    int tn = blockIdx.x % tilesN;
    int row0 = tm * 64, col0 = tn * 64;
    int tid  = threadIdx.x;
    int wave = tid >> 5, lane = tid & 31;
    int l16 = lane & 15, hi = lane >> 4;

    v8f acc[4] = {};

    // A row owned by this lane (M % 64 == 0 -> always in bounds)
    const float* Arow = A + (size_t)(row0 + wave * 16 + l16) * K;

    // W-fill mapping: each thread owns a 4(k) x 4(n) sub-block of the slab
    int nq    = (tid & 15) * 4;       // n offset in tile: 0,4,...,60
    int kb    = (tid >> 4) * 4;       // k offset in slab: 0,4,...,28
    int khalf = kb >> 4;              // 0 or 1
    int kin   = kb & 15;              // 0,4,8,12
    int gn4   = col0 + nq;
    float msk = (gn4 < N) ? 1.f : 0.f;             // value mask (no EXEC branch)
    int   gnc = (gn4 < N) ? gn4 : 0;               // clamped, address stays legal
    const float* Wcol = W + gnc;

    int kFull = K & ~31;
    for (int k0 = 0; k0 < kFull; k0 += 32) {
        // ---- cooperative W slab load (vector, branch-free) -> LDS (bf16) ----
        const float* Wk = Wcol + (size_t)(k0 + kb) * N;
        float4 r0 = *(const float4*)(Wk);
        float4 r1 = *(const float4*)(Wk + (size_t)N);
        float4 r2 = *(const float4*)(Wk + (size_t)2 * N);
        float4 r3 = *(const float4*)(Wk + (size_t)3 * N);
        float c0[4] = { r0.x * msk, r0.y * msk, r0.z * msk, r0.w * msk };
        float c1[4] = { r1.x * msk, r1.y * msk, r1.z * msk, r1.w * msk };
        float c2[4] = { r2.x * msk, r2.y * msk, r2.z * msk, r2.w * msk };
        float c3[4] = { r3.x * msk, r3.y * msk, r3.z * msk, r3.w * msk };
        #pragma unroll
        for (int j = 0; j < 4; j++) {
            union { __bf16 h[4]; uint2 u; } q;
            q.h[0] = (__bf16)c0[j]; q.h[1] = (__bf16)c1[j];
            q.h[2] = (__bf16)c2[j]; q.h[3] = (__bf16)c3[j];
            *(uint2*)&sW[((khalf * 64) + nq + j) * 16 + kin] = q.u;
        }

        // ---- A fragment: e=0..15 -> k0+hi*16+e ----
        const float4* pa0 = (const float4*)(Arow + k0 + hi * 16);
        const float4* pa1 = (const float4*)(Arow + k0 + hi * 16 + 8);
        float4 a0 = pa0[0], a1 = pa0[1];
        float4 a2 = pa1[0], a3 = pa1[1];
        v16bf a;
        a[0]=(__bf16)a0.x;  a[1]=(__bf16)a0.y;  a[2]=(__bf16)a0.z;  a[3]=(__bf16)a0.w;
        a[4]=(__bf16)a1.x;  a[5]=(__bf16)a1.y;  a[6]=(__bf16)a1.z;  a[7]=(__bf16)a1.w;
        a[8]=(__bf16)a2.x;  a[9]=(__bf16)a2.y;  a[10]=(__bf16)a2.z; a[11]=(__bf16)a2.w;
        a[12]=(__bf16)a3.x; a[13]=(__bf16)a3.y; a[14]=(__bf16)a3.z; a[15]=(__bf16)a3.w;

        __syncthreads();
        #pragma unroll
        for (int s = 0; s < 4; s++) {
            v16bf b = *(const v16bf*)&sW[(hi * 64 + s * 16 + l16) * 16];
            acc[s] = __builtin_amdgcn_wmma_f32_16x16x32_bf16(
                        false, a, false, b, (short)0, acc[s], false, false);
        }
        __syncthreads();
    }

    // ---- K remainder (clamped + masked, executed at most once) ----
    if (kFull < K) {
        int k0 = kFull;
        float c0[4], c1[4], c2[4], c3[4];
        {
            int k = k0 + kb;
            int kc0 = (k     < K) ? (k    ) : (K - 1);
            int kc1 = (k + 1 < K) ? (k + 1) : (K - 1);
            int kc2 = (k + 2 < K) ? (k + 2) : (K - 1);
            int kc3 = (k + 3 < K) ? (k + 3) : (K - 1);
            float m0 = ((k     < K) ? 1.f : 0.f) * msk;
            float m1 = ((k + 1 < K) ? 1.f : 0.f) * msk;
            float m2 = ((k + 2 < K) ? 1.f : 0.f) * msk;
            float m3 = ((k + 3 < K) ? 1.f : 0.f) * msk;
            float4 r0 = *(const float4*)(Wcol + (size_t)kc0 * N);
            float4 r1 = *(const float4*)(Wcol + (size_t)kc1 * N);
            float4 r2 = *(const float4*)(Wcol + (size_t)kc2 * N);
            float4 r3 = *(const float4*)(Wcol + (size_t)kc3 * N);
            c0[0]=r0.x*m0; c0[1]=r0.y*m0; c0[2]=r0.z*m0; c0[3]=r0.w*m0;
            c1[0]=r1.x*m1; c1[1]=r1.y*m1; c1[2]=r1.z*m1; c1[3]=r1.w*m1;
            c2[0]=r2.x*m2; c2[1]=r2.y*m2; c2[2]=r2.z*m2; c2[3]=r2.w*m2;
            c3[0]=r3.x*m3; c3[1]=r3.y*m3; c3[2]=r3.z*m3; c3[3]=r3.w*m3;
        }
        #pragma unroll
        for (int j = 0; j < 4; j++) {
            union { __bf16 h[4]; uint2 u; } q;
            q.h[0] = (__bf16)c0[j]; q.h[1] = (__bf16)c1[j];
            q.h[2] = (__bf16)c2[j]; q.h[3] = (__bf16)c3[j];
            *(uint2*)&sW[((khalf * 64) + nq + j) * 16 + kin] = q.u;
        }

        v16bf a;
        #pragma unroll
        for (int e = 0; e < 16; e++) {
            int k = k0 + hi * 16 + e;
            int kc = (k < K) ? k : (K - 1);
            float v = Arow[kc] * ((k < K) ? 1.f : 0.f);
            a[e] = (__bf16)v;
        }

        __syncthreads();
        #pragma unroll
        for (int s = 0; s < 4; s++) {
            v16bf b = *(const v16bf*)&sW[(hi * 64 + s * 16 + l16) * 16];
            acc[s] = __builtin_amdgcn_wmma_f32_16x16x32_bf16(
                        false, a, false, b, (short)0, acc[s], false, false);
        }
    }

    // ---- epilogue: C layout VGPR e -> m = (lane/16)*8 + e, n = lane%16 ----
    int mBase = row0 + wave * 16 + hi * 8;
    #pragma unroll
    for (int s = 0; s < 4; s++) {
        int n = col0 + s * 16 + l16;
        if (n >= N) continue;
        float bv = bias ? bias[n] : 0.f;
        #pragma unroll
        for (int e = 0; e < 8; e++) {
            int m = mBase + e;
            float v = acc[s][e] + bv;
            if (act == ACT_RELU)     v = fmaxf(v, 0.f);
            else if (act == ACT_SIG) v = 1.f / (1.f + __expf(-v));
            if (m < M) C[(size_t)m * ldc + n] = v;
        }
    }
}

// ---------------------------------------------------------------------------
// Residual LayerNorm (block per row; in-place safe: row staged in LDS first)
// ---------------------------------------------------------------------------
__global__ void ln_kernel(const float* __restrict__ a, const float* __restrict__ r,
                          const float* __restrict__ g, const float* __restrict__ be,
                          float* __restrict__ out, int n)
{
    __shared__ float sv[1024];
    __shared__ float red[256];
    int row = blockIdx.x, tid = threadIdx.x;
    size_t base = (size_t)row * n;
    float ls = 0.f;
    for (int i = tid; i < n; i += 256) {
        float v = a[base + i] + (r ? r[base + i] : 0.f);
        sv[i] = v; ls += v;
    }
    red[tid] = ls; __syncthreads();
    for (int s = 128; s > 0; s >>= 1) { if (tid < s) red[tid] += red[tid + s]; __syncthreads(); }
    float mean = red[0] / n;
    __syncthreads();
    float lv = 0.f;
    for (int i = tid; i < n; i += 256) { float d = sv[i] - mean; lv += d * d; }
    red[tid] = lv; __syncthreads();
    for (int s = 128; s > 0; s >>= 1) { if (tid < s) red[tid] += red[tid + s]; __syncthreads(); }
    float rstd = rsqrtf(red[0] / n + 1e-5f);
    for (int i = tid; i < n; i += 256)
        out[base + i] = (sv[i] - mean) * rstd * g[i] + be[i];
}

// ---------------------------------------------------------------------------
// Fused attention: one block per (b, head, query-row). hd <= 256, S <= 128.
// qkv rows are [q | k | v] of width 3d per token.
// ---------------------------------------------------------------------------
__global__ void attn_kernel(const float* __restrict__ qkv, float* __restrict__ out,
                            int B, int S, int d, int nhead)
{
    __shared__ float sQ[256];
    __shared__ float sS[128];
    __shared__ float sInv;
    int hd = d / nhead;
    int id = blockIdx.x;
    int q = id % S; int bh = id / S;
    int h = bh % nhead; int b = bh / nhead;
    int tid = threadIdx.x;
    size_t rowQ = (size_t)(b * S + q) * 3 * d;
    for (int i = tid; i < hd; i += 128) sQ[i] = qkv[rowQ + h * hd + i];
    __syncthreads();
    float scale = rsqrtf((float)hd);
    for (int kk = tid; kk < S; kk += 128) {
        const float* kr = qkv + (size_t)(b * S + kk) * 3 * d + d + h * hd;
        float s = 0.f;
        for (int i = 0; i < hd; i++) s += sQ[i] * kr[i];
        sS[kk] = s * scale;
    }
    __syncthreads();
    if (tid == 0) {
        float mx = -3.4e38f;
        for (int kk = 0; kk < S; kk++) mx = fmaxf(mx, sS[kk]);
        float sum = 0.f;
        for (int kk = 0; kk < S; kk++) { float e = __expf(sS[kk] - mx); sS[kk] = e; sum += e; }
        sInv = 1.f / sum;
    }
    __syncthreads();
    float inv = sInv;
    for (int i = tid; i < hd; i += 128) {
        float o = 0.f;
        for (int kk = 0; kk < S; kk++)
            o += sS[kk] * qkv[(size_t)(b * S + kk) * 3 * d + 2 * d + h * hd + i];
        out[(size_t)(b * S + q) * d + h * hd + i] = o * inv;
    }
}

// ---------------------------------------------------------------------------
// GRU pointwise gate update (after h@W_hh GEMM)
// ---------------------------------------------------------------------------
__global__ void gru_point(const float* __restrict__ xi, long xi_rs,
                          const float* __restrict__ hg,
                          float* __restrict__ h,
                          float* __restrict__ hout, long ho_rs,
                          int B, int H)
{
    int i = blockIdx.x * blockDim.x + threadIdx.x;
    if (i >= B * H) return;
    int b = i / H, j = i % H;
    float xr = xi[(size_t)b * xi_rs + j];
    float xz = xi[(size_t)b * xi_rs + H + j];
    float xn = xi[(size_t)b * xi_rs + 2 * H + j];
    float hr = hg[(size_t)b * 3 * H + j];
    float hz = hg[(size_t)b * 3 * H + H + j];
    float hn = hg[(size_t)b * 3 * H + 2 * H + j];
    float rg = 1.f / (1.f + __expf(-(xr + hr)));
    float zg = 1.f / (1.f + __expf(-(xz + hz)));
    float ng = tanhf(xn + rg * hn);
    float hv = (1.f - zg) * ng + zg * h[(size_t)b * H + j];
    h[(size_t)b * H + j] = hv;
    hout[(size_t)b * ho_rs + j] = hv;
}

// ---------------------------------------------------------------------------
// Misc small kernels
// ---------------------------------------------------------------------------
__global__ void fill_zero(float* p, size_t n)
{
    size_t i = (size_t)blockIdx.x * blockDim.x + threadIdx.x;
    if (i < n) p[i] = 0.f;
}

__global__ void transpose_btd(const float* __restrict__ x, float* __restrict__ xt,
                              int B, int T, int D)
{
    size_t i = (size_t)blockIdx.x * blockDim.x + threadIdx.x;
    size_t tot = (size_t)B * T * D;
    if (i >= tot) return;
    int b = (int)(i / ((size_t)T * D));
    int rem = (int)(i % ((size_t)T * D));
    int t = rem / D, dd = rem % D;
    xt[((size_t)b * D + dd) * T + t] = x[i];
}

__global__ void row_mean(const float* __restrict__ a, float* __restrict__ out, int R, int L)
{
    int r = blockIdx.x * blockDim.x + threadIdx.x;
    if (r >= R) return;
    float s = 0.f;
    for (int i = 0; i < L; i++) s += a[(size_t)r * L + i];
    out[r] = s / (float)L;
}

// spa_tem[b,t,d] = tx[b,t,d]*aS[b,d] + sx[b,d,t]*aT[b,t] + x[b,t,d]
__global__ void combine_kernel(const float* __restrict__ tx, const float* __restrict__ sx,
                               const float* __restrict__ aS, const float* __restrict__ aT,
                               const float* __restrict__ x, float* __restrict__ out,
                               int B, int T, int D)
{
    size_t i = (size_t)blockIdx.x * blockDim.x + threadIdx.x;
    size_t tot = (size_t)B * T * D;
    if (i >= tot) return;
    int b = (int)(i / ((size_t)T * D));
    int rem = (int)(i % ((size_t)T * D));
    int t = rem / D, dd = rem % D;
    out[i] = tx[i] * aS[(size_t)b * D + dd]
           + sx[((size_t)b * D + dd) * T + t] * aT[(size_t)b * T + t]
           + x[i];
}

__global__ void cb_norm_inv(const float* __restrict__ cb, float* __restrict__ inv, int NC, int E)
{
    int c = blockIdx.x * blockDim.x + threadIdx.x;
    if (c >= NC) return;
    float s = 0.f;
    for (int e = 0; e < E; e++) { float v = cb[(size_t)c * E + e]; s += v * v; }
    inv[c] = 1.f / (sqrtf(s) + 1e-12f);
}

// argmax over codes of (z . c) * ||c||^-1  (||z|| row-constant -> irrelevant)
__global__ void vq_argmax(const float* __restrict__ z, const float* __restrict__ cb,
                          const float* __restrict__ cninv, int* __restrict__ idx,
                          int E, int NC)
{
    __shared__ float sz[256];
    __shared__ float sB[256];
    __shared__ int   sI[256];
    int row = blockIdx.x, tid = threadIdx.x;
    for (int i = tid; i < E; i += 256) sz[i] = z[(size_t)row * E + i];
    __syncthreads();
    float best = -3.4e38f; int bi = 0x7fffffff;
    for (int c = tid; c < NC; c += 256) {
        const float* cr = cb + (size_t)c * E;
        float s = 0.f;
        for (int e = 0; e < E; e++) s += sz[e] * cr[e];
        s *= cninv[c];
        if (s > best) { best = s; bi = c; }
    }
    sB[tid] = best; sI[tid] = bi; __syncthreads();
    for (int st = 128; st > 0; st >>= 1) {
        if (tid < st) {
            if (sB[tid + st] > sB[tid] ||
                (sB[tid + st] == sB[tid] && sI[tid + st] < sI[tid])) {
                sB[tid] = sB[tid + st]; sI[tid] = sI[tid + st];
            }
        }
        __syncthreads();
    }
    if (tid == 0) idx[row] = sI[0];
}

__global__ void vq_gather(const float* __restrict__ z, const float* __restrict__ cb,
                          const int* __restrict__ idx, float* __restrict__ zq,
                          float* __restrict__ acc, int E)
{
    __shared__ float red[256];
    int row = blockIdx.x, tid = threadIdx.x;
    int c = idx[row];
    float ls = 0.f;
    for (int i = tid; i < E; i += 256) {
        float q = cb[(size_t)c * E + i];
        float d = q - z[(size_t)row * E + i];
        zq[(size_t)row * E + i] = q;
        ls += d * d;
    }
    red[tid] = ls; __syncthreads();
    for (int s = 128; s > 0; s >>= 1) { if (tid < s) red[tid] += red[tid + s]; __syncthreads(); }
    if (tid == 0) atomicAdd(acc, red[0]);
}

__global__ void vq_finalize(const float* __restrict__ acc, float* __restrict__ out, float scale)
{
    out[0] = acc[0] * scale;   // loss_s = 1.25 * mse
    out[1] = acc[1] * scale;   // loss_t
}

// ---------------------------------------------------------------------------
// Host-side orchestration
// ---------------------------------------------------------------------------
static const int B_ = 512, T_ = 64, DIN = 100, HID_ = 512, EMB_ = 256, NC_ = 512, NH_ = 4;
static const int M1 = B_ * T_;    // 32768 tokens
static const int M2 = B_ * DIN;   // 51200 tokens

struct LayerW { const float *qw,*qb,*ow,*ob,*g1,*b1,*l1w,*l1b,*l2w,*l2b,*g2,*b2; };
struct EncW   { LayerW L[2]; const float *ng,*nb; };

static EncW load_enc(void* const* d_in, int& ii)
{
    EncW e;
    for (int l = 0; l < 2; l++) {
        LayerW& w = e.L[l];
        w.qw  = (const float*)d_in[ii++]; w.qb  = (const float*)d_in[ii++];
        w.ow  = (const float*)d_in[ii++]; w.ob  = (const float*)d_in[ii++];
        w.g1  = (const float*)d_in[ii++]; w.b1  = (const float*)d_in[ii++];
        w.l1w = (const float*)d_in[ii++]; w.l1b = (const float*)d_in[ii++];
        w.l2w = (const float*)d_in[ii++]; w.l2b = (const float*)d_in[ii++];
        w.g2  = (const float*)d_in[ii++]; w.b2  = (const float*)d_in[ii++];
    }
    e.ng = (const float*)d_in[ii++];
    e.nb = (const float*)d_in[ii++];
    return e;
}

// M must be a multiple of 64 and N a multiple of 4 at every call site.
static void gemm(const float* A, const float* W, const float* bias, float* C,
                 int M, int N, int K, int ldc, int act, hipStream_t st)
{
    int tilesM = (M + 63) / 64, tilesN = (N + 63) / 64;
    gemm_wmma_bf16<<<tilesM * tilesN, 128, 0, st>>>(A, W, bias, C, M, N, K, ldc, act, tilesN);
}

static void run_encoder(float* cur, const EncW& E, int Bb, int S, int d, int ff, int nhead,
                        float* qkvB, float* attnB, float* hidB, float* tmpB, hipStream_t st)
{
    int M = Bb * S;
    for (int l = 0; l < 2; l++) {
        const LayerW& w = E.L[l];
        gemm(cur, w.qw, w.qb, qkvB, M, 3 * d, d, 3 * d, ACT_NONE, st);
        attn_kernel<<<Bb * nhead * S, 128, 0, st>>>(qkvB, attnB, Bb, S, d, nhead);
        gemm(attnB, w.ow, w.ob, tmpB, M, d, d, d, ACT_NONE, st);
        ln_kernel<<<M, 256, 0, st>>>(cur, tmpB, w.g1, w.b1, cur, d);
        gemm(cur, w.l1w, w.l1b, hidB, M, ff, d, ff, ACT_RELU, st);
        gemm(hidB, w.l2w, w.l2b, tmpB, M, d, ff, d, ACT_NONE, st);
        ln_kernel<<<M, 256, 0, st>>>(cur, tmpB, w.g2, w.b2, cur, d);
    }
    ln_kernel<<<M, 256, 0, st>>>(cur, nullptr, E.ng, E.nb, cur, d);
}

static void run_gru(const float* xin, int din,
                    const float* wih, const float* whh, const float* bih, const float* bhh,
                    float* xi, float* h, float* hg, float* hseq, hipStream_t st)
{
    // Precompute input gates for all timesteps: [M1, 3H]
    gemm(xin, wih, bih, xi, M1, 3 * HID_, din, 3 * HID_, ACT_NONE, st);
    fill_zero<<<(B_ * HID_ + 255) / 256, 256, 0, st>>>(h, (size_t)B_ * HID_);
    for (int t = 0; t < T_; t++) {
        // batch-parallel recurrence: [512,512] @ [512,1536] per step (WMMA)
        gemm(h, whh, bhh, hg, B_, 3 * HID_, HID_, 3 * HID_, ACT_NONE, st);
        gru_point<<<(B_ * HID_ + 255) / 256, 256, 0, st>>>(
            xi + (size_t)t * 3 * HID_, (long)T_ * 3 * HID_,
            hg, h, hseq + (size_t)t * HID_, (long)T_ * HID_, B_, HID_);
    }
}

extern "C" void kernel_launch(void* const* d_in, const int* in_sizes, int n_in,
                              void* d_out, int out_size, void* d_ws, size_t ws_size,
                              hipStream_t stream)
{
    (void)in_sizes; (void)n_in; (void)out_size; (void)ws_size;

    // ---- unpack inputs in setup_inputs() flatten order ----
    int ii = 0;
    const float* x = (const float*)d_in[ii++];
    EncW spaE = load_enc(d_in, ii);                       // spatial_enc  (d=100)
    EncW temE = load_enc(d_in, ii);                       // temporal_enc (d=64)
    const float* sa_l1w = (const float*)d_in[ii++]; const float* sa_l1b = (const float*)d_in[ii++];
    const float* sa_l2w = (const float*)d_in[ii++]; const float* sa_l2b = (const float*)d_in[ii++];
    const float* ta_l1w = (const float*)d_in[ii++]; const float* ta_l1b = (const float*)d_in[ii++];
    const float* ta_l2w = (const float*)d_in[ii++]; const float* ta_l2b = (const float*)d_in[ii++];
    const float* g0_wih = (const float*)d_in[ii++]; const float* g0_whh = (const float*)d_in[ii++];
    const float* g0_bih = (const float*)d_in[ii++]; const float* g0_bhh = (const float*)d_in[ii++];
    const float* g1_wih = (const float*)d_in[ii++]; const float* g1_whh = (const float*)d_in[ii++];
    const float* g1_bih = (const float*)d_in[ii++]; const float* g1_bhh = (const float*)d_in[ii++];
    const float* sp_w  = (const float*)d_in[ii++]; const float* sp_b  = (const float*)d_in[ii++];
    const float* sfe_w = (const float*)d_in[ii++]; const float* sfe_b = (const float*)d_in[ii++];
    const float* tfe_w = (const float*)d_in[ii++]; const float* tfe_b = (const float*)d_in[ii++];
    const float* cb_s  = (const float*)d_in[ii++];
    const float* cb_t  = (const float*)d_in[ii++];
    const float* sfd_w = (const float*)d_in[ii++]; const float* sfd_b = (const float*)d_in[ii++];
    const float* tfd_w = (const float*)d_in[ii++]; const float* tfd_b = (const float*)d_in[ii++];
    EncW decE = load_enc(d_in, ii);                       // decoder (d=1024)
    const float* pr_w = (const float*)d_in[ii++]; const float* pr_b = (const float*)d_in[ii++];

    // ---- workspace layout (64-float aligned so float4 loads stay aligned) ----
    float* wsp = (float*)d_ws;
    size_t off = 0;
    auto alloc = [&](size_t n) { float* p = wsp + off; off += (n + 63) & ~(size_t)63; return p; };
    float* tx     = alloc((size_t)M1 * DIN);        // spatial-enc activations [B,T,100]
    float* sx     = alloc((size_t)M2 * T_);         // temporal-enc activations [B,100,64]
    float* meanT  = alloc((size_t)M1);              // [B,64]
    float* meanS  = alloc((size_t)M2);              // [B,100]
    float* hid8   = alloc((size_t)B_ * 8);
    float* attnT  = alloc((size_t)B_ * T_);         // temporal_attn [B,64]
    float* attnS  = alloc((size_t)B_ * DIN);        // spatial_attn  [B,100]
    float* spa    = alloc((size_t)M1 * DIN);        // spa_tem
    float* xi     = alloc((size_t)M1 * 3 * HID_);   // GRU input gates
    float* h      = alloc((size_t)B_ * HID_);       // GRU hidden
    float* hg     = alloc((size_t)B_ * 3 * HID_);   // GRU recurrent gates
    float* trans0 = alloc((size_t)M1 * HID_);
    float* trans1 = alloc((size_t)M1 * HID_);
    float* stateH = alloc((size_t)M1 * HID_);
    float* states = alloc((size_t)M1 * EMB_);
    float* transE = alloc((size_t)M1 * EMB_);
    float* qs     = alloc((size_t)M1 * EMB_);
    float* qt     = alloc((size_t)M1 * EMB_);
    float* cninv  = alloc((size_t)NC_);
    float* idxf   = alloc((size_t)M1);
    float* lossA  = alloc((size_t)2);
    float* dec    = alloc((size_t)M1 * 2 * HID_);   // concat buffer [B,T,1024]
    // shared encoder scratch, sized for the d=1024 decoder (covers all three)
    float* s_qkv  = alloc((size_t)M1 * 3072);
    float* s_attn = alloc((size_t)M1 * 1024);
    float* s_hid  = alloc((size_t)M1 * 1024);
    float* s_tmp  = alloc((size_t)M1 * 1024);
    int*   vqidx  = (int*)idxf;
    float* out    = (float*)d_out;

    // ---- 0. zero loss accumulators (replay-safe) ----
    fill_zero<<<1, 64, 0, stream>>>(lossA, 2);

    // ---- 1. spatial encoder on a copy of x ----
    hipMemcpyAsync(tx, x, sizeof(float) * (size_t)M1 * DIN, hipMemcpyDeviceToDevice, stream);
    run_encoder(tx, spaE, B_, T_, DIN, 512, NH_, s_qkv, s_attn, s_hid, s_tmp, stream);

    // ---- 2. temporal encoder on x^T ----
    {
        size_t tot = (size_t)M1 * DIN;
        transpose_btd<<<(int)((tot + 255) / 256), 256, 0, stream>>>(x, sx, B_, T_, DIN);
    }
    run_encoder(sx, temE, B_, DIN, T_, 512, NH_, s_qkv, s_attn, s_hid, s_tmp, stream);

    // ---- 3. gating MLPs ----
    row_mean<<<(M1 + 255) / 256, 256, 0, stream>>>(tx, meanT, M1, DIN);   // [B,64]
    row_mean<<<(M2 + 255) / 256, 256, 0, stream>>>(sx, meanS, M2, T_);    // [B,100]
    gemm(meanT, sa_l1w, sa_l1b, hid8, B_, 8, T_, 8, ACT_RELU, stream);
    gemm(hid8,  sa_l2w, sa_l2b, attnT, B_, T_, 8, T_, ACT_SIG, stream);
    gemm(meanS, ta_l1w, ta_l1b, hid8, B_, 8, DIN, 8, ACT_RELU, stream);
    gemm(hid8,  ta_l2w, ta_l2b, attnS, B_, DIN, 8, DIN, ACT_SIG, stream);

    // ---- 4. spa_tem combine ----
    {
        size_t tot = (size_t)M1 * DIN;
        combine_kernel<<<(int)((tot + 255) / 256), 256, 0, stream>>>(
            tx, sx, attnS, attnT, x, spa, B_, T_, DIN);
    }

    // ---- 5. two GRU layers (batch-parallel recurrence via WMMA) ----
    run_gru(spa,    DIN,  g0_wih, g0_whh, g0_bih, g0_bhh, xi, h, hg, trans0, stream);
    run_gru(trans0, HID_, g1_wih, g1_whh, g1_bih, g1_bhh, xi, h, hg, trans1, stream);

    // ---- 6. projections ----
    gemm(spa,    sp_w,  sp_b,  stateH, M1, HID_, DIN,  HID_, ACT_NONE, stream);
    gemm(stateH, sfe_w, sfe_b, states, M1, EMB_, HID_, EMB_, ACT_NONE, stream);
    gemm(trans1, tfe_w, tfe_b, transE, M1, EMB_, HID_, EMB_, ACT_NONE, stream);

    // ---- 7. vector quantization (cosine argmax + straight-through + loss) ----
    cb_norm_inv<<<(NC_ + 255) / 256, 256, 0, stream>>>(cb_s, cninv, NC_, EMB_);
    vq_argmax<<<M1, 256, 0, stream>>>(states, cb_s, cninv, vqidx, EMB_, NC_);
    vq_gather<<<M1, 256, 0, stream>>>(states, cb_s, vqidx, qs, lossA + 0, EMB_);
    cb_norm_inv<<<(NC_ + 255) / 256, 256, 0, stream>>>(cb_t, cninv, NC_, EMB_);
    vq_argmax<<<M1, 256, 0, stream>>>(transE, cb_t, cninv, vqidx, EMB_, NC_);
    vq_gather<<<M1, 256, 0, stream>>>(transE, cb_t, vqidx, qt, lossA + 1, EMB_);

    // ---- 8. decoder: concat via ldc-strided GEMMs, then d=1024 encoder ----
    gemm(qs, sfd_w, sfd_b, dec,        M1, HID_, EMB_, 2 * HID_, ACT_NONE, stream);
    gemm(qt, tfd_w, tfd_b, dec + HID_, M1, HID_, EMB_, 2 * HID_, ACT_NONE, stream);
    run_encoder(dec, decE, B_, T_, 2 * HID_, 2 * HID_, NH_, s_qkv, s_attn, s_hid, s_tmp, stream);

    // ---- 9. prediction head + loss scalars ----
    gemm(dec, pr_w, pr_b, out, M1, DIN, 2 * HID_, DIN, ACT_NONE, stream);
    vq_finalize<<<1, 1, 0, stream>>>(lossA, out + (size_t)M1 * DIN,
                                     (1.0f + 0.25f) / ((float)M1 * (float)EMB_));
}
